// NearestNeighborAffineContour_4183298146932
// MI455X (gfx1250) — matrix operand order repaired and checked by reading, not compiled, not run on hardware
//
#include <hip/hip_runtime.h>

// CDNA5 / gfx1250: wave32, WMMA 16x16x32 f16 -> f32
typedef _Float16 v16h __attribute__((ext_vector_type(16)));
typedef _Float16 v8h  __attribute__((ext_vector_type(8)));
typedef float    v8f  __attribute__((ext_vector_type(8)));

#define HALF_SITES 1048576
#define WAVES      8
#define NTILES     4
#define SITES_PER_BLOCK (WAVES * NTILES * 16)          // 512
#define NBLOCKS    (HALF_SITES / SITES_PER_BLOCK)      // 2048

union AFrag { v16h v; v8h h[2]; };

__device__ __forceinline__ float bcast_lane(float v, int srclane) {
  return __int_as_float(__builtin_amdgcn_readlane(__float_as_int(v), srclane));
}

__global__ __launch_bounds__(256)
void contour_wmma_kernel(const float* __restrict__ x,
                         const int*   __restrict__ nn_idx,
                         const float* __restrict__ W1u, const float* __restrict__ b1u,
                         const float* __restrict__ W2u, const float* __restrict__ b2u,
                         const float* __restrict__ W1v, const float* __restrict__ b1v,
                         const float* __restrict__ W2v, const float* __restrict__ b2v,
                         const float* __restrict__ Wsu, const float* __restrict__ bsu,
                         const float* __restrict__ Wtv, const float* __restrict__ btv,
                         float* __restrict__ out)
{
  // W2 staged transposed (n-major, f16) so each B fragment is 32 contiguous bytes
  __shared__ _Float16 sW2u[64 * 64];
  __shared__ _Float16 sW2v[64 * 64];
  __shared__ _Float16 sH1u[WAVES][16 * 64];   // per-wave layer-1 activations (A-tile)
  __shared__ _Float16 sH1v[WAVES][16 * 64];
  __shared__ float    sUS[WAVES][16];
  __shared__ float    sUT[WAVES][16];

  const int tid  = threadIdx.x;
  const int wave = tid >> 5;
  const int lane = tid & 31;
  const int g    = lane >> 4;   // half-wave group
  const int nn   = lane & 15;   // N / M position within 16

  // ---- one-time: W2 f32 -> f16 transpose into LDS (coalesced global reads) ----
  for (int i = tid; i < 4096; i += 256) {
    int k = i >> 6, n = i & 63;
    sW2u[n * 64 + k] = (_Float16)W2u[i];
    sW2v[n * 64 + k] = (_Float16)W2v[i];
  }
  __syncthreads();

  // ---- hoist ALL loop-invariant B fragments (W2) into registers ----
  // B fragment (16x16x32 f16): lane = (g,n); element e -> k = c*32 + g*16 + e
  v16h Bu[4][2], Bv[4][2];
#pragma unroll
  for (int t = 0; t < 4; ++t) {
#pragma unroll
    for (int c = 0; c < 2; ++c) {
      Bu[t][c] = *(const v16h*)(sW2u + (t * 16 + nn) * 64 + c * 32 + g * 16);
      Bv[t][c] = *(const v16h*)(sW2v + (t * 16 + nn) * 64 + c * 32 + g * 16);
    }
  }

  // ---- persistent per-lane weights ----
  // layer-1: lane owns output features f = 2*lane, 2*lane+1
  float w1u0[5], w1u1[5], w1v0[5], w1v1[5];
#pragma unroll
  for (int j = 0; j < 5; ++j) {
    float2 wu = *(const float2*)(W1u + j * 64 + 2 * lane);
    float2 wv = *(const float2*)(W1v + j * 64 + 2 * lane);
    w1u0[j] = wu.x; w1u1[j] = wu.y;
    w1v0[j] = wv.x; w1v1[j] = wv.y;
  }
  const float2 bu2 = *(const float2*)(b1u + 2 * lane);
  const float2 bv2 = *(const float2*)(b1v + 2 * lane);
  float b2u_r[4], b2v_r[4], ws_r[4], wt_r[4];
#pragma unroll
  for (int t = 0; t < 4; ++t) {
    b2u_r[t] = b2u[t * 16 + nn];
    b2v_r[t] = b2v[t * 16 + nn];
    ws_r[t]  = Wsu[t * 16 + nn];
    wt_r[t]  = Wtv[t * 16 + nn];
  }
  const float bs = bsu[0];
  const float bt = btv[0];

  const int base0 = (blockIdx.x * WAVES + wave) * (NTILES * 16);
  _Float16* const hu = &sH1u[wave][0];
  _Float16* const hv = &sH1v[wave][0];

#pragma unroll 1
  for (int tile = 0; tile < NTILES; ++tile) {
    const int sbase = base0 + tile * 16;

    // ---- irregular gather: lanes 0..15 own one site each (x is L2-resident) ----
    float xv0 = 0.f, xv1 = 0.f, xv2 = 0.f, xv3 = 0.f, xv4 = 0.f;
    if (lane < 16) {
      const int* r = nn_idx + (size_t)(sbase + lane) * 5;
      xv0 = x[r[0]]; xv1 = x[r[1]]; xv2 = x[r[2]]; xv3 = x[r[3]]; xv4 = x[r[4]];
    }

    // ---- layer 1 (K=5, VALU): lane computes features 2L,2L+1 for all 16 sites ----
#pragma unroll
    for (int m = 0; m < 16; ++m) {
      const float s0 = bcast_lane(xv0, m);
      const float s1 = bcast_lane(xv1, m);
      const float s2 = bcast_lane(xv2, m);
      const float s3 = bcast_lane(xv3, m);
      const float s4 = bcast_lane(xv4, m);

      float h0 = bu2.x, h1 = bu2.y, h2 = bv2.x, h3 = bv2.y;
      h0 = fmaf(s0, w1u0[0], h0); h1 = fmaf(s0, w1u1[0], h1);
      h2 = fmaf(s0, w1v0[0], h2); h3 = fmaf(s0, w1v1[0], h3);
      h0 = fmaf(s1, w1u0[1], h0); h1 = fmaf(s1, w1u1[1], h1);
      h2 = fmaf(s1, w1v0[1], h2); h3 = fmaf(s1, w1v1[1], h3);
      h0 = fmaf(s2, w1u0[2], h0); h1 = fmaf(s2, w1u1[2], h1);
      h2 = fmaf(s2, w1v0[2], h2); h3 = fmaf(s2, w1v1[2], h3);
      h0 = fmaf(s3, w1u0[3], h0); h1 = fmaf(s3, w1u1[3], h1);
      h2 = fmaf(s3, w1v0[3], h2); h3 = fmaf(s3, w1v1[3], h3);
      h0 = fmaf(s4, w1u0[4], h0); h1 = fmaf(s4, w1u1[4], h1);
      h2 = fmaf(s4, w1v0[4], h2); h3 = fmaf(s4, w1v1[4], h3);
      h0 = fmaxf(h0, 0.f); h1 = fmaxf(h1, 0.f);
      h2 = fmaxf(h2, 0.f); h3 = fmaxf(h3, 0.f);

      union { _Float16 f[2]; unsigned u; } pu, pv;
      pu.f[0] = (_Float16)h0; pu.f[1] = (_Float16)h1;
      pv.f[0] = (_Float16)h2; pv.f[1] = (_Float16)h3;
      *(unsigned*)(hu + m * 64 + 2 * lane) = pu.u;
      *(unsigned*)(hv + m * 64 + 2 * lane) = pv.u;
    }

    // cross-lane LDS RAW within the wave: force DS completion
    asm volatile("s_wait_dscnt 0" ::: "memory");

    // ---- A fragments: contiguous k = g*16 + e, same order as B (b128 loads) ----
    AFrag au0, au1, av0, av1;
    {
      const _Float16* pu = hu + nn * 64;
      const _Float16* pv = hv + nn * 64;
      au0.h[0] = *(const v8h*)(pu + g * 16);       // K 0..7   / 16..23
      au0.h[1] = *(const v8h*)(pu + 8 + g * 16);   // K 8..15  / 24..31
      au1.h[0] = *(const v8h*)(pu + 32 + g * 16);  // K chunk 1
      au1.h[1] = *(const v8h*)(pu + 40 + g * 16);
      av0.h[0] = *(const v8h*)(pv + g * 16);
      av0.h[1] = *(const v8h*)(pv + 8 + g * 16);
      av1.h[0] = *(const v8h*)(pv + 32 + g * 16);
      av1.h[1] = *(const v8h*)(pv + 40 + g * 16);
    }

    // ---- layer 2 + heads: 16 WMMAs, fold bias+ReLU+head per column tile ----
    // keeps only cu/cv (16 VGPRs) + su/sv (16 VGPRs) live -> stays <256 VGPRs
    float su[8], sv[8];
#pragma unroll
    for (int r = 0; r < 8; ++r) { su[r] = 0.f; sv[r] = 0.f; }

#pragma unroll
    for (int t = 0; t < 4; ++t) {
      v8f cu = {}; v8f cv = {};
      cu = __builtin_amdgcn_wmma_f32_16x16x32_f16(false, au0.v, false, Bu[t][0], (short)0, cu, false, false);
      cv = __builtin_amdgcn_wmma_f32_16x16x32_f16(false, av0.v, false, Bv[t][0], (short)0, cv, false, false);
      cu = __builtin_amdgcn_wmma_f32_16x16x32_f16(false, au1.v, false, Bu[t][1], (short)0, cu, false, false);
      cv = __builtin_amdgcn_wmma_f32_16x16x32_f16(false, av1.v, false, Bv[t][1], (short)0, cv, false, false);
#pragma unroll
      for (int r = 0; r < 8; ++r) {   // C row m = 8*g + r
        su[r] = fmaf(fmaxf(cu[r] + b2u_r[t], 0.f), ws_r[t], su[r]);
        sv[r] = fmaf(fmaxf(cv[r] + b2v_r[t], 0.f), wt_r[t], sv[r]);
      }
    }

    // ---- reduce across the 16-lane half (wave32-safe xor masks < 16) ----
#pragma unroll
    for (int r = 0; r < 8; ++r) {
#pragma unroll
      for (int mask = 8; mask >= 1; mask >>= 1) {
        su[r] += __shfl_xor(su[r], mask, 32);
        sv[r] += __shfl_xor(sv[r], mask, 32);
      }
    }
    if (nn == 0) {
#pragma unroll
      for (int r = 0; r < 8; ++r) {
        sUS[wave][8 * g + r] = su[r] + bs;
        sUT[wave][8 * g + r] = sv[r] + bt;
      }
    }
    asm volatile("s_wait_dscnt 0" ::: "memory");

    // ---- final: one float4 per site covers full interleaved complex output ----
    if (lane < 16) {
      const int site = sbase + lane;
      const float us = sUS[wave][lane];
      const float ut = sUT[wave][lane];
      const float2 x2 = *(const float2*)(x + 2 * (size_t)site);  // x[2s], x[2s+1]
      float4 o;
      o.x = x2.x;                  // Re z[2s]   = x[2s]
      o.y = 0.f;                   // Im z[2s]   = 0
      o.z = x2.y;                  // Re z[2s+1] = x[2s+1]
      o.w = fmaf(us, x2.y, ut);    // Im z[2s+1] = u_s*x_odd + u_t
      *(float4*)(out + 4 * (size_t)site) = o;
    }
  }
}

extern "C" void kernel_launch(void* const* d_in, const int* in_sizes, int n_in,
                              void* d_out, int out_size, void* d_ws, size_t ws_size,
                              hipStream_t stream) {
  (void)in_sizes; (void)n_in; (void)d_ws; (void)ws_size; (void)out_size;
  const float* x      = (const float*)d_in[0];
  const int*   nn_idx = (const int*)d_in[1];
  // d_in[2] = odd_indices (implicitly 2j+1; recomputed in-kernel)
  const float* W1u = (const float*)d_in[3];
  const float* b1u = (const float*)d_in[4];
  const float* W2u = (const float*)d_in[5];
  const float* b2u = (const float*)d_in[6];
  const float* W1v = (const float*)d_in[7];
  const float* b1v = (const float*)d_in[8];
  const float* W2v = (const float*)d_in[9];
  const float* b2v = (const float*)d_in[10];
  const float* Wsu = (const float*)d_in[11];
  const float* bsu = (const float*)d_in[12];
  const float* Wtv = (const float*)d_in[13];
  const float* btv = (const float*)d_in[14];

  hipLaunchKernelGGL(contour_wmma_kernel, dim3(NBLOCKS), dim3(256), 0, stream,
                     x, nn_idx, W1u, b1u, W2u, b2u, W1v, b1v, W2v, b2v,
                     Wsu, bsu, Wtv, btv, (float*)d_out);
}
